// TensorCoreAttentionV2_67972152426567
// MI455X (gfx1250) — matrix operand order repaired and checked by reading, not compiled, not run on hardware
//
#include <hip/hip_runtime.h>

typedef __attribute__((ext_vector_type(16))) __bf16 v16bf;
typedef __attribute__((ext_vector_type(8)))  float  v8f;
typedef __attribute__((ext_vector_type(4)))  int    v4i;
typedef unsigned short u16;
typedef unsigned int   u32;

#if defined(__AMDGCN__) && __has_builtin(__builtin_amdgcn_global_load_async_to_lds_b128)
#define USE_ASYNC 1
#else
#define USE_ASYNC 0
#endif

struct Frag { union { u32 u[8]; v16bf v; }; };

static __device__ __forceinline__ v8f wmma_bf16(const Frag& a, const Frag& b, v8f c) {
  // D = A(16x32 bf16) x B(32x16 bf16) + C(16x16 f32)
  return __builtin_amdgcn_wmma_f32_16x16x32_bf16(false, a.v, false, b.v, (short)0, c, false, false);
}

// Async global -> LDS 16-byte copy (CDNA5 GLOBAL_LOAD_ASYNC_TO_LDS_B128,
// tracked by ASYNCcnt). Falls back to a VGPR-staged copy when the builtin is
// not declared by this toolchain.
static __device__ __forceinline__ void async_copy16(const u16* g, u16* l) {
#if USE_ASYNC
  __builtin_amdgcn_global_load_async_to_lds_b128(
      (__attribute__((address_space(1))) v4i*)(u16*)g,
      (__attribute__((address_space(3))) v4i*)l, 0, 0);
#else
  *reinterpret_cast<uint4*>(l) = *reinterpret_cast<const uint4*>(g);
#endif
}

static __device__ __forceinline__ void async_wait() {
#if USE_ASYNC
#if __has_builtin(__builtin_amdgcn_s_wait_asynccnt)
  __builtin_amdgcn_s_wait_asynccnt(0);
#else
  asm volatile("s_wait_asynccnt 0" ::: "memory");
#endif
#endif
}

static __device__ __forceinline__ u16 f2bf(float f) {
  u32 u = __float_as_uint(f);
  u32 r = (u + 0x7FFFu + ((u >> 16) & 1u)) >> 16;  // round-to-nearest-even
  return (u16)r;
}

static __device__ __forceinline__ float expapx(float x, const float* c) {
  float acc = c[0], p = 1.f;
#pragma unroll
  for (int i = 1; i < 7; ++i) { p *= x; acc = fmaf(c[i], p, acc); }
  return fmaxf(acc, 1e-8f);
}

static __device__ __forceinline__ float gelu_poly(float x, const float* c) {
  x = fminf(fmaxf(x, -4.f), 4.f);
  float acc = c[0], p = 1.f;
#pragma unroll
  for (int i = 1; i < 8; ++i) { p *= x; acc = fmaf(c[i], p, acc); }
  return acc;
}

// ---------------------------------------------------------------------------
// f32 -> bf16 conversion
// ---------------------------------------------------------------------------
__global__ void cvt_bf16_kernel(const float* __restrict__ s, u16* __restrict__ d, int n) {
  int i = blockIdx.x * blockDim.x + threadIdx.x;
  int stride = gridDim.x * blockDim.x;
  for (; i < n; i += stride) d[i] = f2bf(s[i]);
}

// ---------------------------------------------------------------------------
// bf16 GEMM: C[MxN] = A[MxK] @ B[KxN]; A,B row-major bf16; f32 accumulate.
// Block tile 128x128, K-step 64, double-buffered software pipeline:
//   iter t: issue async A(t+1) + global B(t+1) loads, compute on buffer t,
//           scatter B(t+1) into LDS, wait asynccnt, one barrier.
// 256 threads = 8 waves (2x4); each wave owns 64x32 = 4x2 accumulators
// -> 16 WMMAs per K-step between barriers.
// MODE 0: f32 out. MODE 1: bf16 out, scaled. MODE 2: gelu-poly -> bf16.
// ---------------------------------------------------------------------------
template <int MODE>
__global__ __launch_bounds__(256) void gemm_bf16_kernel(
    const u16* __restrict__ A, const u16* __restrict__ Bm,
    float* __restrict__ Cf, u16* __restrict__ Cb,
    int M, int N, int K, float scale, const float* __restrict__ gelu_c)
{
  __shared__ u16 a_tile[2][128][72];   // row-major [m][k]
  __shared__ u16 b_tile[2][128][72];   // transposed [n][k]

  const int tid  = threadIdx.x;
  const int lane = tid & 31;
  const int wave = tid >> 5;
  const int wm   = wave >> 2;   // 0..1
  const int wn   = wave & 3;    // 0..3
  const int half = lane >> 4;
  const int ln16 = lane & 15;
  const int bm = blockIdx.y, bn = blockIdx.x;

  const int arow = tid >> 3, acol = (tid & 7) * 8;     // A loader coords
  const int bkrow = tid >> 4, bcol = (tid & 15) * 8;   // B loader coords
  uint4 breg[4];

  auto issueA = [&](int kk, int buf) {
#pragma unroll
    for (int p = 0; p < 4; ++p) {
      int row = p * 32 + arow;
      async_copy16(A + (size_t)(bm * 128 + row) * K + kk + acol, &a_tile[buf][row][acol]);
    }
  };
  auto loadB = [&](int kk) {
#pragma unroll
    for (int p = 0; p < 4; ++p)
      breg[p] = *reinterpret_cast<const uint4*>(Bm + (size_t)(kk + p * 16 + bkrow) * N + bn * 128 + bcol);
  };
  auto storeB = [&](int buf) {
#pragma unroll
    for (int p = 0; p < 4; ++p) {
      union { uint4 q; u16 s[8]; } uv; uv.q = breg[p];
#pragma unroll
      for (int j = 0; j < 8; ++j) b_tile[buf][bcol + j][p * 16 + bkrow] = uv.s[j];
    }
  };

  float gc[8];
  if constexpr (MODE == 2) {
#pragma unroll
    for (int i = 0; i < 8; ++i) gc[i] = gelu_c[i];
  }

  v8f acc[4][2];
#pragma unroll
  for (int mt = 0; mt < 4; ++mt)
#pragma unroll
    for (int nt = 0; nt < 2; ++nt)
#pragma unroll
      for (int i = 0; i < 8; ++i) acc[mt][nt][i] = 0.0f;

  // pipeline prologue: fill buffer 0
  issueA(0, 0);
  loadB(0);
  storeB(0);
  async_wait();
  __syncthreads();

  const int nk = K / 64;
  for (int t = 0; t < nk; ++t) {
    const int buf = t & 1, nbuf = buf ^ 1;
    const bool more = (t + 1) < nk;
    if (more) { issueA((t + 1) * 64, nbuf); loadB((t + 1) * 64); }

#pragma unroll
    for (int kc = 0; kc < 64; kc += 32) {
      Frag afr[4], bfr[2];
#pragma unroll
      for (int mt = 0; mt < 4; ++mt) {
        int rowb = wm * 64 + mt * 16 + ln16;
#pragma unroll
        for (int v = 0; v < 8; ++v) {
          int k0 = (v < 4) ? (2 * v + 8 * half) : (16 + 2 * (v - 4) + 8 * half);
          afr[mt].u[v] = *reinterpret_cast<const u32*>(&a_tile[buf][rowb][kc + k0]);
        }
      }
#pragma unroll
      for (int nt = 0; nt < 2; ++nt) {
        int nb = wn * 32 + nt * 16 + ln16;
#pragma unroll
        for (int v = 0; v < 8; ++v) {
          int k0 = (v < 4) ? (2 * v + 8 * half) : (16 + 2 * (v - 4) + 8 * half);
          bfr[nt].u[v] = *reinterpret_cast<const u32*>(&b_tile[buf][nb][kc + k0]);
        }
      }
#pragma unroll
      for (int mt = 0; mt < 4; ++mt)
#pragma unroll
        for (int nt = 0; nt < 2; ++nt)
          acc[mt][nt] = wmma_bf16(afr[mt], bfr[nt], acc[mt][nt]);
    }

    if (more) { storeB(nbuf); async_wait(); }
    __syncthreads();
  }

  // epilogue: C layout — lane ln16 = N, VGPR r = M row (r + 8*half)
#pragma unroll
  for (int mt = 0; mt < 4; ++mt) {
#pragma unroll
    for (int nt = 0; nt < 2; ++nt) {
#pragma unroll
      for (int r = 0; r < 8; ++r) {
        int row = bm * 128 + wm * 64 + mt * 16 + r + 8 * half;
        int col = bn * 128 + wn * 32 + nt * 16 + ln16;
        float v = acc[mt][nt][r];
        if constexpr (MODE == 0)      Cf[(size_t)row * N + col] = v;
        else if constexpr (MODE == 1) Cb[(size_t)row * N + col] = f2bf(v * scale);
        else                          Cb[(size_t)row * N + col] = f2bf(gelu_poly(v, gc));
      }
    }
  }
}

// ---------------------------------------------------------------------------
// Flash-style attention. One block per (q-block of 128, head, batch).
// Q pre-scaled by 1/sqrt(DH) at projection. Streams 64-key K/V tiles with a
// double-buffered pipeline: async K(t+1) + V(t+1) register loads are issued
// before the S/softmax/PV work on buffer t. One barrier per iteration.
// Online softmax with reference polynomial exp (max-subtracted, clamp 1e-8).
// ---------------------------------------------------------------------------
__global__ __launch_bounds__(256) void attn_kernel(
    const u16* __restrict__ Q, const u16* __restrict__ Kt,
    const u16* __restrict__ V, u16* __restrict__ O,
    const float* __restrict__ exp_c)
{
  __shared__ u16 q_tile[128][72];      // [q][dh]
  __shared__ u16 k_tile[2][64][72];    // [key][dh]  (B operand for S = Q K^T)
  __shared__ u16 v_t   [2][64][72];    // [dh][key]  (B operand for O = P V)
  __shared__ u16 p_tile[128][72];      // [q][key]   (A operand for O = P V)

  const int tid  = threadIdx.x;
  const int lane = tid & 31, wave = tid >> 5;
  const int half = lane >> 4, ln16 = lane & 15;
  const int qb = blockIdx.x, h = blockIdx.y, b = blockIdx.z;
  const size_t rowbase = (size_t)b * 2048 + (size_t)qb * 128;
  const int colbase = h * 64;

  const int lrow = tid >> 3, lcol = (tid & 7) * 8;
  uint4 vreg[2];

  auto issueK = [&](int kt, int buf) {
#pragma unroll
    for (int p = 0; p < 2; ++p) {
      int row = p * 32 + lrow;
      async_copy16(Kt + ((size_t)b * 2048 + kt * 64 + row) * 512 + colbase + lcol,
                   &k_tile[buf][row][lcol]);
    }
  };
  auto loadV = [&](int kt) {
#pragma unroll
    for (int p = 0; p < 2; ++p)
      vreg[p] = *reinterpret_cast<const uint4*>(
          V + ((size_t)b * 2048 + kt * 64 + p * 32 + lrow) * 512 + colbase + lcol);
  };
  auto storeV = [&](int buf) {
#pragma unroll
    for (int p = 0; p < 2; ++p) {
      union { uint4 q; u16 s[8]; } uv; uv.q = vreg[p];
#pragma unroll
      for (int j = 0; j < 8; ++j) v_t[buf][lcol + j][p * 32 + lrow] = uv.s[j];
    }
  };

  float ec[7];
#pragma unroll
  for (int i = 0; i < 7; ++i) ec[i] = exp_c[i];

  // prologue: Q tile 128x64 (async) + first K/V tile
#pragma unroll
  for (int p = 0; p < 4; ++p) {
    int row = p * 32 + lrow;
    async_copy16(Q + (rowbase + row) * 512 + colbase + lcol, &q_tile[row][lcol]);
  }
  issueK(0, 0);
  loadV(0);
  storeV(0);
  async_wait();
  __syncthreads();

  v8f oacc[4];
  float mrow[8], lrow_sm[8];
#pragma unroll
  for (int dt = 0; dt < 4; ++dt)
#pragma unroll
    for (int i = 0; i < 8; ++i) oacc[dt][i] = 0.f;
#pragma unroll
  for (int r = 0; r < 8; ++r) { mrow[r] = -1e30f; lrow_sm[r] = 0.f; }

  const int qr0 = wave * 16;

  for (int kt = 0; kt < 32; ++kt) {
    const int buf = kt & 1, nbuf = buf ^ 1;
    const bool more = (kt + 1) < 32;
    if (more) { issueK(kt + 1, nbuf); loadV(kt + 1); }

    // S = Q K^T : wave computes 16 q-rows x 64 keys
    v8f sacc[4];
#pragma unroll
    for (int nt = 0; nt < 4; ++nt)
#pragma unroll
      for (int i = 0; i < 8; ++i) sacc[nt][i] = 0.f;
#pragma unroll
    for (int kc = 0; kc < 64; kc += 32) {
      Frag af;
#pragma unroll
      for (int v = 0; v < 8; ++v) {
        int k0 = (v < 4) ? (2 * v + 8 * half) : (16 + 2 * (v - 4) + 8 * half);
        af.u[v] = *reinterpret_cast<const u32*>(&q_tile[qr0 + ln16][kc + k0]);
      }
#pragma unroll
      for (int nt = 0; nt < 4; ++nt) {
        Frag bf;
#pragma unroll
        for (int v = 0; v < 8; ++v) {
          int k0 = (v < 4) ? (2 * v + 8 * half) : (16 + 2 * (v - 4) + 8 * half);
          bf.u[v] = *reinterpret_cast<const u32*>(&k_tile[buf][nt * 16 + ln16][kc + k0]);
        }
        sacc[nt] = wmma_bf16(af, bf, sacc[nt]);
      }
    }

    // online softmax update (row = r + 8*half within wave's 16 rows)
    float tmax[8];
#pragma unroll
    for (int r = 0; r < 8; ++r) {
      float t = sacc[0][r];
#pragma unroll
      for (int nt = 1; nt < 4; ++nt) t = fmaxf(t, sacc[nt][r]);
#pragma unroll
      for (int m = 1; m < 16; m <<= 1) t = fmaxf(t, __shfl_xor(t, m, 32));
      tmax[r] = t;
    }
    float sc[8], rsum[8];
#pragma unroll
    for (int r = 0; r < 8; ++r) {
      float mnew = fmaxf(mrow[r], tmax[r]);
      float d = mrow[r] - mnew;
      sc[r] = (d < -30.f) ? 0.f : expapx(d, ec);
      mrow[r] = mnew;
      rsum[r] = 0.f;
    }
#pragma unroll
    for (int nt = 0; nt < 4; ++nt) {
#pragma unroll
      for (int r = 0; r < 8; ++r) {
        float e = expapx(sacc[nt][r] - mrow[r], ec);
        rsum[r] += e;
        p_tile[qr0 + r + 8 * half][nt * 16 + ln16] = f2bf(e);
      }
    }
#pragma unroll
    for (int r = 0; r < 8; ++r) {
#pragma unroll
      for (int m = 1; m < 16; m <<= 1) rsum[r] += __shfl_xor(rsum[r], m, 32);
      lrow_sm[r] = lrow_sm[r] * sc[r] + rsum[r];
    }
#pragma unroll
    for (int dt = 0; dt < 4; ++dt)
#pragma unroll
      for (int i = 0; i < 8; ++i) oacc[dt][i] *= sc[i];

    // O += P V  (wave reads only its own p_tile rows; no barrier needed)
#pragma unroll
    for (int kc = 0; kc < 64; kc += 32) {
      Frag af;
#pragma unroll
      for (int v = 0; v < 8; ++v) {
        int k0 = (v < 4) ? (2 * v + 8 * half) : (16 + 2 * (v - 4) + 8 * half);
        af.u[v] = *reinterpret_cast<const u32*>(&p_tile[qr0 + ln16][kc + k0]);
      }
#pragma unroll
      for (int dt = 0; dt < 4; ++dt) {
        Frag bf;
#pragma unroll
        for (int v = 0; v < 8; ++v) {
          int k0 = (v < 4) ? (2 * v + 8 * half) : (16 + 2 * (v - 4) + 8 * half);
          bf.u[v] = *reinterpret_cast<const u32*>(&v_t[buf][dt * 16 + ln16][kc + k0]);
        }
        oacc[dt] = wmma_bf16(af, bf, oacc[dt]);
      }
    }

    if (more) { storeV(nbuf); async_wait(); }
    __syncthreads();
  }

  // normalize & store bf16 [B*T, D] at col = h*64 + dh
#pragma unroll
  for (int r = 0; r < 8; ++r) {
    float inv = 1.0f / lrow_sm[r];
    size_t row = rowbase + qr0 + r + 8 * half;
#pragma unroll
    for (int dt = 0; dt < 4; ++dt)
      O[row * 512 + colbase + dt * 16 + ln16] = f2bf(oacc[dt][r] * inv);
  }
}

// ---------------------------------------------------------------------------
// LayerNorm with residual: out = LN(a + b); matches reference Newton rsqrt.
// One block (256 threads) per 512-wide row.
// ---------------------------------------------------------------------------
__global__ __launch_bounds__(256) void ln_kernel(
    const float* __restrict__ a, const float* __restrict__ b,
    const float* __restrict__ gamma, const float* __restrict__ beta,
    float* __restrict__ outf, u16* __restrict__ outb)
{
  __shared__ float rs[256], rq[256];
  const int tid = threadIdx.x;
  const size_t row = blockIdx.x;
  float v0 = a[row * 512 + tid] + b[row * 512 + tid];
  float v1 = a[row * 512 + tid + 256] + b[row * 512 + tid + 256];
  rs[tid] = v0 + v1;
  rq[tid] = v0 * v0 + v1 * v1;
  __syncthreads();
  for (int s = 128; s > 0; s >>= 1) {
    if (tid < s) { rs[tid] += rs[tid + s]; rq[tid] += rq[tid + s]; }
    __syncthreads();
  }
  float mean = rs[0] * (1.0f / 512.0f);
  float var  = rq[0] * (1.0f / 512.0f) - mean * mean;
  float vv = var + 1e-5f;
  float y = 0.5f;
#pragma unroll
  for (int i = 0; i < 3; ++i) y = y * (3.0f - vv * y * y) * 0.5f;
  float o0 = (v0 - mean) * y * gamma[tid] + beta[tid];
  float o1 = (v1 - mean) * y * gamma[tid + 256] + beta[tid + 256];
  outf[row * 512 + tid] = o0;
  outf[row * 512 + tid + 256] = o1;
  if (outb) {
    outb[row * 512 + tid] = f2bf(o0);
    outb[row * 512 + tid + 256] = f2bf(o1);
  }
}

// ---------------------------------------------------------------------------
// Host launcher
// ---------------------------------------------------------------------------
extern "C" void kernel_launch(void* const* d_in, const int* in_sizes, int n_in,
                              void* d_out, int out_size, void* d_ws, size_t ws_size,
                              hipStream_t stream)
{
  (void)in_sizes; (void)n_in; (void)out_size; (void)ws_size;
  const int Bsz = 4, T = 2048, D = 512, D4 = 2048;
  const int BT = Bsz * T;                 // 8192

  const float* x      = (const float*)d_in[0];
  const float* Wq     = (const float*)d_in[1];
  const float* Wk     = (const float*)d_in[2];
  const float* Wv     = (const float*)d_in[3];
  const float* Wo     = (const float*)d_in[4];
  const float* W1     = (const float*)d_in[5];
  const float* W2     = (const float*)d_in[6];
  const float* g1     = (const float*)d_in[7];
  const float* b1     = (const float*)d_in[8];
  const float* g2     = (const float*)d_in[9];
  const float* b2     = (const float*)d_in[10];
  const float* gelu_c = (const float*)d_in[11];
  const float* exp_c  = (const float*)d_in[12];
  float* out = (float*)d_out;

  char* ws = (char*)d_ws;
  size_t off = 0;
  auto carve = [&](size_t bytes) -> void* {
    void* p = ws + off;
    off += (bytes + 255) & ~(size_t)255;
    return p;
  };

  u16*   xb    = (u16*)  carve((size_t)BT * D * 2);
  u16*   wqb   = (u16*)  carve((size_t)D * D * 2);
  u16*   wkb   = (u16*)  carve((size_t)D * D * 2);
  u16*   wvb   = (u16*)  carve((size_t)D * D * 2);
  u16*   wob   = (u16*)  carve((size_t)D * D * 2);
  u16*   w1b   = (u16*)  carve((size_t)D * D4 * 2);
  u16*   w2b   = (u16*)  carve((size_t)D4 * D * 2);
  u16*   qbuf  = (u16*)  carve((size_t)BT * D * 2);
  u16*   kbuf  = (u16*)  carve((size_t)BT * D * 2);
  u16*   vbuf  = (u16*)  carve((size_t)BT * D * 2);
  u16*   abuf  = (u16*)  carve((size_t)BT * D * 2);
  float* proj  = (float*)carve((size_t)BT * D * 4);
  float* hbuf  = (float*)carve((size_t)BT * D * 4);
  u16*   hb    = (u16*)  carve((size_t)BT * D * 2);
  u16*   gact  = (u16*)  carve((size_t)BT * D4 * 2);
  float* ffn   = (float*)carve((size_t)BT * D * 4);

  // 1) f32 -> bf16
  cvt_bf16_kernel<<<1024, 256, 0, stream>>>(x,  xb,  BT * D);
  cvt_bf16_kernel<<<256,  256, 0, stream>>>(Wq, wqb, D * D);
  cvt_bf16_kernel<<<256,  256, 0, stream>>>(Wk, wkb, D * D);
  cvt_bf16_kernel<<<256,  256, 0, stream>>>(Wv, wvb, D * D);
  cvt_bf16_kernel<<<256,  256, 0, stream>>>(Wo, wob, D * D);
  cvt_bf16_kernel<<<512,  256, 0, stream>>>(W1, w1b, D * D4);
  cvt_bf16_kernel<<<512,  256, 0, stream>>>(W2, w2b, D4 * D);

  // 2) Q,K,V projections (Q pre-scaled by 1/sqrt(64))
  dim3 gqkv(D / 128, BT / 128);
  gemm_bf16_kernel<1><<<gqkv, 256, 0, stream>>>(xb, wqb, nullptr, qbuf, BT, D, D, 0.125f, nullptr);
  gemm_bf16_kernel<1><<<gqkv, 256, 0, stream>>>(xb, wkb, nullptr, kbuf, BT, D, D, 1.0f, nullptr);
  gemm_bf16_kernel<1><<<gqkv, 256, 0, stream>>>(xb, wvb, nullptr, vbuf, BT, D, D, 1.0f, nullptr);

  // 3) attention
  attn_kernel<<<dim3(T / 128, 8, Bsz), 256, 0, stream>>>(qbuf, kbuf, vbuf, abuf, exp_c);

  // 4) output projection + residual LN
  gemm_bf16_kernel<0><<<gqkv, 256, 0, stream>>>(abuf, wob, proj, nullptr, BT, D, D, 1.0f, nullptr);
  ln_kernel<<<BT, 256, 0, stream>>>(x, proj, g1, b1, hbuf, hb);

  // 5) FFN: gelu(h @ W1) @ W2, residual LN -> output
  gemm_bf16_kernel<2><<<dim3(D4 / 128, BT / 128), 256, 0, stream>>>(hb, w1b, nullptr, gact, BT, D4, D, 1.0f, gelu_c);
  gemm_bf16_kernel<0><<<gqkv, 256, 0, stream>>>(gact, w2b, ffn, nullptr, BT, D, D4, 1.0f, nullptr);
  ln_kernel<<<BT, 256, 0, stream>>>(hbuf, ffn, g2, b2, out, nullptr);
}
